// NodeFeatLayer_79517024518209
// MI455X (gfx1250) — matrix operand order repaired and checked by reading, not compiled, not run on hardware
//
#include <hip/hip_runtime.h>
#include <hip/hip_bf16.h>
#include <math.h>

// Problem constants (match reference setup_inputs)
#define BB   64
#define NN   100
#define KK   16
#define CC   1024
#define RR   8
#define DD   128
#define OUTC 1024
#define BN   (BB*NN)          // 6400
#define EE   (BN*KK)          // 102400

typedef __attribute__((ext_vector_type(16))) __bf16 v16bf;
typedef __attribute__((ext_vector_type(8)))  float  v8f;

__device__ __forceinline__ unsigned short f32_to_bf16_rne(float f) {
    unsigned int u = __float_as_uint(f);
    if ((u & 0x7F800000u) == 0x7F800000u) {          // inf / nan: truncate
        return (unsigned short)(u >> 16);
    }
    unsigned int r = u + 0x7FFFu + ((u >> 16) & 1u); // round to nearest even
    return (unsigned short)(r >> 16);
}

// ---------------------------------------------------------------------------
// f32 -> bf16 bulk convert
// ---------------------------------------------------------------------------
__global__ void k_cvt_bf16(const float* __restrict__ in,
                           unsigned short* __restrict__ out, int n) {
    int i = blockIdx.x * 256 + threadIdx.x;
    if (i < n) out[i] = f32_to_bf16_rne(in[i]);
}

// ---------------------------------------------------------------------------
// conv_w [R][C][D] f32  ->  Wt [R*D][C] bf16 (TRANSPOSED weight):
//   Wt[j][c] = conv_w[r, c, d]  with j = r*D + d
// Row j of Wt is contiguous in the GEMM K dimension (c), so B tiles can be
// DMA'd straight to LDS with async b128 transfers (no in-kernel transpose).
// ---------------------------------------------------------------------------
__global__ void k_pack_wt(const float* __restrict__ cw,
                          unsigned short* __restrict__ out) {
    int o = blockIdx.x * 256 + threadIdx.x;           // over (R*D)*C = 1M
    if (o >= RR * DD * CC) return;
    int c = o & (CC - 1);
    int j = o >> 10;                                   // /C
    int r = j >> 7;                                    // /D
    int d = j & (DD - 1);
    out[o] = f32_to_bf16_rne(cw[((size_t)r * CC + c) * DD + d]);
}

// ---------------------------------------------------------------------------
// Edge geometric weights for BOTH layers.
//   w2 = normalized gaussian-mixture weights (layer2 params)
//   w1 = graph_w * normalized weights (layer1 params)
// ---------------------------------------------------------------------------
__global__ void k_edge_weights(const float* __restrict__ centre,   // [BN][2]
                               const float* __restrict__ graph_w,  // [E]
                               const int*   __restrict__ nbr_idx,  // [E]
                               const float* __restrict__ mr1, const float* __restrict__ mt1,
                               const float* __restrict__ pr1, const float* __restrict__ pt1,
                               const float* __restrict__ mr2, const float* __restrict__ mt2,
                               const float* __restrict__ pr2, const float* __restrict__ pt2,
                               float* __restrict__ w1,             // [E][R]
                               float* __restrict__ w2) {           // [E][R]
    int e = blockIdx.x * 256 + threadIdx.x;
    if (e >= EE) return;
    const float TWO_PI = 6.283185307179586f;
    int bn = e / KK;
    int b  = bn / NN;
    int nb = b * NN + nbr_idx[e];
    float dx = centre[2 * bn]     - centre[2 * nb];
    float dy = centre[2 * bn + 1] - centre[2 * nb + 1];
    float rho   = sqrtf(dx * dx + dy * dy);
    float theta = atan2f(dx, dy);                      // arctan2(d0, d1)

    float wa[RR], wb[RR];
    float sa = 0.f, sb = 0.f;
    #pragma unroll
    for (int r = 0; r < RR; ++r) {
        {
            float dr = rho - mr1[r];
            float wr = __expf(-0.5f * dr * dr / (1e-14f + pr1[r] * pr1[r]));
            float fa = fabsf(theta - mt1[r]);
            float an = fminf(fa, fabsf(TWO_PI - fa));
            float wt = __expf(-0.5f * an * an / (1e-14f + pt1[r] * pt1[r]));
            float w  = wr * wt;
            if (w != w) w = 0.f;                       // nan -> 0
            wa[r] = w; sa += w;
        }
        {
            float dr = rho - mr2[r];
            float wr = __expf(-0.5f * dr * dr / (1e-14f + pr2[r] * pr2[r]));
            float fa = fabsf(theta - mt2[r]);
            float an = fminf(fa, fabsf(TWO_PI - fa));
            float wt = __expf(-0.5f * an * an / (1e-14f + pt2[r] * pt2[r]));
            float w  = wr * wt;
            if (w != w) w = 0.f;
            wb[r] = w; sb += w;
        }
    }
    float gw  = graph_w[e];
    float isa = 1.f / sa;
    float isb = 1.f / sb;
    #pragma unroll
    for (int r = 0; r < RR; ++r) {
        w1[(size_t)e * RR + r] = gw * wa[r] * isa;
        w2[(size_t)e * RR + r] = wb[r] * isb;
    }
}

// ---------------------------------------------------------------------------
// bf16 WMMA GEMM:  P[M][Ncol] = A[M][Kd] (bf16) @ Wt[Ncol][Kd]^T (bf16), f32 out
// Block tile 128x128, 8 waves (4 rows x 2 cols), wave tile 32x64 = 2x4 wmma
// tiles, KT=32.  Tiles staged memory->LDS with GLOBAL_LOAD_ASYNC_TO_LDS_B128
// (ASYNCcnt), double-buffered so tile t+1's DMA overlaps tile t's WMMAs.
// ---------------------------------------------------------------------------
#define TM  128
#define TN  128
#define TK  32
#define LDA 40   // LDS row stride (bf16 elems); 80 bytes = 5*16 -> b128 aligned
#define LDB 40

__global__ __launch_bounds__(256) void k_gemm_bf16(
        const unsigned short* __restrict__ A,   // [M][Kd]    bf16 bits, row-major
        const unsigned short* __restrict__ Wt,  // [Ncol][Kd] bf16 bits, row-major (W^T)
        float* __restrict__ P,                  // [M][Ncol]  f32
        int M, int Ncol, int Kd) {
    __shared__ __align__(16) unsigned short As[2 * TM * LDA];
    __shared__ __align__(16) unsigned short Bs[2 * TN * LDB];

    const int tid  = threadIdx.x;
    const int lane = tid & 31;
    const int wave = tid >> 5;
    const int wm   = wave & 3;           // wave row (0..3) -> 32 rows each
    const int wn   = wave >> 2;          // wave col (0..1) -> 64 cols each
    const int m0   = blockIdx.x * TM;
    const int n0   = blockIdx.y * TN;
    const int half = lane >> 4;          // lane group (0: lanes 0-15, 1: 16-31)
    const int l16  = lane & 15;

    // Issue one K-tile's DMA into LDS buffer `buf`.
    // Per wave: 4 async-load instructions -> ASYNCcnt += 4 per tile.
    auto issue_tile = [&](int t, int buf) {
        const int k0 = t * TK;
        unsigned abase = (unsigned)(size_t)(&As[buf * (TM * LDA)]);
        unsigned bbase = (unsigned)(size_t)(&Bs[buf * (TN * LDB)]);
        #pragma unroll
        for (int i = 0; i < 2; ++i) {                   // A tile: 512 b128, 2/thread
            int idx = tid + 256 * i;
            int row = idx >> 2;                          // 0..127
            int seg = idx & 3;                           // 16-byte segment
            const unsigned short* g = A + (size_t)(m0 + row) * Kd + k0 + seg * 8;
            unsigned l = abase + (unsigned)(row * LDA + seg * 8) * 2;
            asm volatile("global_load_async_to_lds_b128 %0, %1, off"
                         :: "v"(l), "v"(g) : "memory");
        }
        #pragma unroll
        for (int i = 0; i < 2; ++i) {                   // B tile: 512 b128, 2/thread
            int idx = tid + 256 * i;
            int row = idx >> 2;                          // 0..127
            int seg = idx & 3;
            const unsigned short* g = Wt + (size_t)(n0 + row) * Kd + k0 + seg * 8;
            unsigned l = bbase + (unsigned)(row * LDB + seg * 8) * 2;
            asm volatile("global_load_async_to_lds_b128 %0, %1, off"
                         :: "v"(l), "v"(g) : "memory");
        }
    };

    v8f acc[2][4];
    #pragma unroll
    for (int ti = 0; ti < 2; ++ti)
        #pragma unroll
        for (int tj = 0; tj < 4; ++tj)
            acc[ti][tj] = (v8f){};

    const int T = Kd / TK;
    issue_tile(0, 0);
    if (T > 1) issue_tile(1, 1);

    for (int t = 0; t < T; ++t) {
        // Async loads complete in order: <=4 outstanding means tile t landed.
        if (t + 1 < T) asm volatile("s_wait_asynccnt 0x4" ::: "memory");
        else           asm volatile("s_wait_asynccnt 0x0" ::: "memory");
        __syncthreads();                      // all waves' tile-t data visible

        const unsigned short* Asb = &As[(t & 1) * (TM * LDA)];
        const unsigned short* Bsb = &Bs[(t & 1) * (TN * LDB)];

        // A 16x32 bf16 layout (ISA 7.12.2): vgpr i<4 : k = 2i        + 8*half
        //                                   vgpr i>=4: k = 16+2(i-4) + 8*half
        // B 32x16 bf16 layout: same k order as A (vgpr i<4: k = 2i + 8*half; i>=4: k = 16+2(i-4) + 8*half)
        union { v16bf v; unsigned int u[8]; } af[2], bf[4];
        #pragma unroll
        for (int i = 0; i < 8; ++i) {
            int ka = (i < 4 ? 2 * i : 16 + 2 * (i - 4)) + 8 * half;
            int kb = (i < 4 ? 2 * i : 16 + 2 * (i - 4)) + 8 * half;
            #pragma unroll
            for (int ti = 0; ti < 2; ++ti)
                af[ti].u[i] = *(const unsigned int*)(&Asb[(wm * 32 + ti * 16 + l16) * LDA + ka]);
            #pragma unroll
            for (int tj = 0; tj < 4; ++tj)
                bf[tj].u[i] = *(const unsigned int*)(&Bsb[(wn * 64 + tj * 16 + l16) * LDB + kb]);
        }
        #pragma unroll
        for (int ti = 0; ti < 2; ++ti)
            #pragma unroll
            for (int tj = 0; tj < 4; ++tj)
                acc[ti][tj] = __builtin_amdgcn_wmma_f32_16x16x32_bf16(
                    false, af[ti].v, false, bf[tj].v, (short)0, acc[ti][tj], false, false);

        __syncthreads();                      // all waves done reading buf (t&1)
        if (t + 2 < T) issue_tile(t + 2, t & 1);
    }

    // store: C/D layout (ISA 7.12.2): vgpr v -> M = v + 8*half, N = l16
    #pragma unroll
    for (int ti = 0; ti < 2; ++ti)
        #pragma unroll
        for (int tj = 0; tj < 4; ++tj)
            #pragma unroll
            for (int v = 0; v < 8; ++v) {
                int gr = m0 + wm * 32 + ti * 16 + v + 8 * half;
                int gc = n0 + wn * 64 + tj * 16 + l16;
                P[(size_t)gr * Ncol + gc] = acc[ti][tj][v];
            }
}

// ---------------------------------------------------------------------------
// Gather: out[bn, r*D+d] = relu( sum_k w[bn,k,r] * P[nbr(bn,k), r*D+d] )
// One block per node. OUT_BF16: emit bf16 (feeds next GEMM) else f32.
// ---------------------------------------------------------------------------
template <bool OUT_BF16>
__global__ __launch_bounds__(256) void k_gather(
        const float* __restrict__ P,        // [BN][1024]
        const float* __restrict__ W,        // [BN][K][R]
        const int*   __restrict__ nbr_idx,  // [BN][K]
        void* __restrict__ out) {
    __shared__ float wsm[KK * RR];          // 128
    __shared__ int   ns[KK];
    const int bn  = blockIdx.x;
    const int tid = threadIdx.x;
    if (tid < KK)      ns[tid]  = (bn / NN) * NN + nbr_idx[(size_t)bn * KK + tid];
    if (tid < KK * RR) wsm[tid] = W[(size_t)bn * KK * RR + tid];
    __syncthreads();
    #pragma unroll
    for (int p = 0; p < 4; ++p) {
        int j = tid + 256 * p;               // 0..1023
        int r = j >> 7;                      // /D
        float acc = 0.f;
        #pragma unroll
        for (int k = 0; k < KK; ++k)
            acc = fmaf(wsm[k * RR + r], P[(size_t)ns[k] * OUTC + j], acc);
        acc = fmaxf(acc, 0.f);
        if (OUT_BF16)
            ((unsigned short*)out)[(size_t)bn * OUTC + j] = f32_to_bf16_rne(acc);
        else
            ((float*)out)[(size_t)bn * OUTC + j] = acc;
    }
}

// ---------------------------------------------------------------------------
extern "C" void kernel_launch(void* const* d_in, const int* in_sizes, int n_in,
                              void* d_out, int out_size, void* d_ws, size_t ws_size,
                              hipStream_t stream) {
    (void)in_sizes; (void)n_in; (void)out_size; (void)ws_size;
    const float* node_feats  = (const float*)d_in[0];   // [B][N][C]
    const float* node_centre = (const float*)d_in[1];   // [B][N][2]
    const float* graph_w     = (const float*)d_in[2];   // [E]
    const int*   nbr_idx     = (const int*)  d_in[3];   // [B][N][K]
    const float* mr1 = (const float*)d_in[4];
    const float* mt1 = (const float*)d_in[5];
    const float* pr1 = (const float*)d_in[6];
    const float* pt1 = (const float*)d_in[7];
    const float* cw1 = (const float*)d_in[8];           // [R][C][D]
    const float* mr2 = (const float*)d_in[9];
    const float* mt2 = (const float*)d_in[10];
    const float* pr2 = (const float*)d_in[11];
    const float* pt2 = (const float*)d_in[12];
    const float* cw2 = (const float*)d_in[13];          // [R][OUT][D]

    // workspace carve-up
    char* ws = (char*)d_ws;
    size_t off = 0;
    auto carve = [&](size_t bytes) { void* p = ws + off; off = (off + bytes + 255) & ~(size_t)255; return p; };
    unsigned short* Xbf  = (unsigned short*)carve((size_t)BN * CC * 2);        // 13.1 MB
    unsigned short* Hbf  = (unsigned short*)carve((size_t)BN * OUTC * 2);      // 13.1 MB
    unsigned short* W1t  = (unsigned short*)carve((size_t)RR * DD * CC * 2);   //  2.1 MB
    unsigned short* W2t  = (unsigned short*)carve((size_t)RR * DD * OUTC * 2); //  2.1 MB
    float*          Pbuf = (float*)carve((size_t)BN * OUTC * 4);               // 26.2 MB
    float*          w1b  = (float*)carve((size_t)EE * RR * 4);                 //  3.3 MB
    float*          w2b  = (float*)carve((size_t)EE * RR * 4);                 //  3.3 MB

    // 1) precision / layout conversions
    {
        int n = BN * CC;
        k_cvt_bf16<<<(n + 255) / 256, 256, 0, stream>>>(node_feats, Xbf, n);
        int nw = RR * DD * CC;
        k_pack_wt<<<(nw + 255) / 256, 256, 0, stream>>>(cw1, W1t);
        k_pack_wt<<<(nw + 255) / 256, 256, 0, stream>>>(cw2, W2t);
    }
    // 2) edge weights (both layers)
    k_edge_weights<<<(EE + 255) / 256, 256, 0, stream>>>(
        node_centre, graph_w, nbr_idx,
        mr1, mt1, pr1, pt1, mr2, mt2, pr2, pt2, w1b, w2b);

    dim3 gemm_grid(BN / TM, OUTC / TN);     // (50, 8)

    // 3) layer 1: P = Xbf @ W1t^T ; gather -> Hbf (bf16, relu)
    k_gemm_bf16<<<gemm_grid, 256, 0, stream>>>(Xbf, W1t, Pbuf, BN, OUTC, CC);
    k_gather<true><<<BN, 256, 0, stream>>>(Pbuf, w1b, nbr_idx, (void*)Hbf);

    // 4) layer 2: P = Hbf @ W2t^T ; gather -> d_out (f32, relu)
    k_gemm_bf16<<<gemm_grid, 256, 0, stream>>>(Hbf, W2t, Pbuf, BN, OUTC, OUTC);
    k_gather<false><<<BN, 256, 0, stream>>>(Pbuf, w2b, nbr_idx, d_out);
}
